// RNNDoubleStacked_53025666236823
// MI455X (gfx1250) — matrix operand order repaired and checked by reading, not compiled, not run on hardware
//
#include <hip/hip_runtime.h>

typedef __attribute__((ext_vector_type(16))) _Float16 v16h;
typedef __attribute__((ext_vector_type(8)))  _Float16 v8h;
typedef __attribute__((ext_vector_type(8)))  float    v8f;

#define PN 4096
#define FN 128
#define HN 128
#define SH 136   // padded LDS row stride in halves (272B) to stagger banks

__device__ __forceinline__ v16h cat8(v8h lo, v8h hi) {
  return __builtin_shufflevector(lo, hi, 0,1,2,3,4,5,6,7,8,9,10,11,12,13,14,15);
}

// Native CDNA5 transcendental tanh (1 TRANS op vs ~40 VALU for libm tanhf).
__device__ __forceinline__ float fast_tanh(float x) {
#if __has_builtin(__builtin_amdgcn_tanhf)
  return __builtin_amdgcn_tanhf(x);
#else
  float y;
  // v_nop guards the TRANS-op result hazard (1 independent op required).
  asm volatile("v_tanh_f32 %0, %1\n\tv_nop" : "=v"(y) : "v"(x));
  return y;
#endif
}

// net1 (batched ragged tanh-RNN, 16 particles / block, WMMA recurrent GEMM)
// + fused net2 input projection: U[p][n] = b2[n] + sum_k h1[p][k]*Wih2[n][k]
__global__ __launch_bounds__(256) void rnn1_kernel(
    const float* __restrict__ event, const int* __restrict__ lengths,
    const float* __restrict__ Wih1,  const float* __restrict__ Whh1,
    const float* __restrict__ bih1,  const float* __restrict__ bhh1,
    const float* __restrict__ Wih2,  const float* __restrict__ bih2,
    const float* __restrict__ bhh2,  float* __restrict__ U)
{
  __shared__ __align__(16) _Float16 hb[2][16][SH];  // double-buffered h (f16)
  __shared__ float xbuf[16][FN];                    // staged inputs

  const int tid  = threadIdx.x;
  const int lane = tid & 31;
  const int wv   = tid >> 5;               // 0..7 : which 16-wide N-tile this wave owns
  const int p0   = blockIdx.x * 16;

  // ---- stage inputs + zero h0 (only buffer 0 is read at t=0) ----
  for (int idx = tid; idx < 16 * FN; idx += 256) {
    int m = idx >> 7, t = idx & (FN - 1);
    xbuf[m][t] = event[(p0 + m) * FN + t];
  }
  for (int idx = tid; idx < 16 * SH; idx += 256)
    ((_Float16*)hb[0])[idx] = (_Float16)0.0f;

  // ---- per-lane constants (C/D layout: VGPR r, lane l -> M=r+8*(l/16), N=l%16) ----
  const int mlo  = (lane >> 4) << 3;       // 0 or 8
  const int ng   = wv * 16 + (lane & 15);  // global hidden index handled by this lane
  const int kbA  = (lane >> 4) << 4;       // A-frag K-half select (0/16)
  const int kbB  = (lane >> 4) << 4;       // B-frag K-half select (0/16)

  const float wih   = Wih1[ng];
  const float bsum1 = bih1[ng] + bhh1[ng];
  const float bsum2 = bih2[ng] + bhh2[ng];

  int lenr[8];
  #pragma unroll
  for (int r = 0; r < 8; ++r) lenr[r] = lengths[p0 + mlo + r];

  // ---- B fragments, register-resident for the whole time loop ----
  // B[k][n] = W[n][k]  (h @ W^T); lane's 16 K-values are contiguous in row n.
  v16h B1[4], B2[4];
  #pragma unroll
  for (int kk = 0; kk < 4; ++kk) {
    const float* w1 = Whh1 + ng * HN + kk * 32 + kbB;
    const float* w2 = Wih2 + ng * HN + kk * 32 + kbB;
    v16h b1, b2;
    #pragma unroll
    for (int i = 0; i < 16; ++i) { b1[i] = (_Float16)w1[i]; b2[i] = (_Float16)w2[i]; }
    B1[kk] = b1; B2[kk] = b2;
  }

  // Per-lane copy of its 8 owned h elements: avoids LDS read-back for the
  // ragged-length freeze (select in registers, store unconditionally).
  float hprev[8];
  #pragma unroll
  for (int r = 0; r < 8; ++r) hprev[r] = 0.0f;

  __syncthreads();

  // ---- recurrence: h <- tanh(x*Wih1 + b + h @ Whh1^T), masked by length ----
  // Double buffer: read hb[par], write hb[par^1]; ONE barrier per step.
  auto step = [&](int t, const _Float16 (*curb)[SH], _Float16 (*nxtb)[SH]) {
    v16h A[4];
    #pragma unroll
    for (int kk = 0; kk < 4; ++kk) {
      const v8h* lo = (const v8h*)&curb[lane & 15][kk * 32 + kbA];
      const v8h* hi = (const v8h*)&curb[lane & 15][kk * 32 + 8 + kbA];
      A[kk] = cat8(*lo, *hi);
    }
    v8f acc = {};
    #pragma unroll
    for (int kk = 0; kk < 4; ++kk)
      acc = __builtin_amdgcn_wmma_f32_16x16x32_f16(false, A[kk], false, B1[kk],
                                                   (short)0, acc, false, false);
    #pragma unroll
    for (int r = 0; r < 8; ++r) {
      const int m = mlo + r;
      float hn  = fast_tanh(acc[r] + bsum1 + wih * xbuf[m][t]);
      float val = (t < lenr[r]) ? hn : hprev[r];   // freeze past ragged length
      hprev[r]  = val;
      nxtb[m][ng] = (_Float16)val;
    }
    __syncthreads();
  };

  for (int t = 0; t < FN; t += 2) {
    step(t,     hb[0], hb[1]);
    step(t + 1, hb[1], hb[0]);
  }
  // FN even -> final h1 lives in hb[0], made visible by the trailing barrier.

  // ---- fused net2 input projection: U tile = h1 @ Wih2^T + (b_ih2+b_hh2) ----
  v16h A[4];
  #pragma unroll
  for (int kk = 0; kk < 4; ++kk) {
    const v8h* lo = (const v8h*)&hb[0][lane & 15][kk * 32 + kbA];
    const v8h* hi = (const v8h*)&hb[0][lane & 15][kk * 32 + 8 + kbA];
    A[kk] = cat8(*lo, *hi);
  }
  v8f acc = {};
  #pragma unroll
  for (int kk = 0; kk < 4; ++kk)
    acc = __builtin_amdgcn_wmma_f32_16x16x32_f16(false, A[kk], false, B2[kk],
                                                 (short)0, acc, false, false);
  #pragma unroll
  for (int r = 0; r < 8; ++r)
    U[(p0 + mlo + r) * HN + ng] = acc[r] + bsum2;
}

// net2 serial scan: h <- tanh(U[p] + Whh2 @ h), 4096 steps, single workgroup.
// Whh2 row fully register-resident per thread; h broadcast from LDS as float4.
__global__ __launch_bounds__(128) void rnn2_kernel(
    const float* __restrict__ U, const float* __restrict__ Whh2,
    float* __restrict__ out)
{
  const int n = threadIdx.x;
  float w[HN];
  #pragma unroll
  for (int k = 0; k < HN; ++k) w[k] = Whh2[n * HN + k];

  __shared__ __align__(16) float hs[HN];
  hs[n] = 0.0f;
  __syncthreads();

  for (int p = 0; p < PN; ++p) {
    float acc = U[p * HN + n];
    if (p + 1 < PN) __builtin_prefetch(&U[(p + 1) * HN + n], 0, 0);
    #pragma unroll
    for (int k = 0; k < HN; k += 4) {            // ds_load_b128 broadcasts
      float4 h4 = *(const float4*)&hs[k];
      acc += w[k] * h4.x + w[k + 1] * h4.y + w[k + 2] * h4.z + w[k + 3] * h4.w;
    }
    float hn = fast_tanh(acc);
    __syncthreads();
    hs[n] = hn;
    __syncthreads();
  }
  out[n] = hs[n];
}

extern "C" void kernel_launch(void* const* d_in, const int* in_sizes, int n_in,
                              void* d_out, int out_size, void* d_ws, size_t ws_size,
                              hipStream_t stream) {
  const float* event   = (const float*)d_in[0];
  const int*   lengths = (const int*)  d_in[1];
  const float* Wih1    = (const float*)d_in[2];
  const float* Whh1    = (const float*)d_in[3];
  const float* bih1    = (const float*)d_in[4];
  const float* bhh1    = (const float*)d_in[5];
  const float* Wih2    = (const float*)d_in[6];
  const float* Whh2    = (const float*)d_in[7];
  const float* bih2    = (const float*)d_in[8];
  const float* bhh2    = (const float*)d_in[9];

  float* U = (float*)d_ws;   // [PN][HN] f32 = 2 MB scratch

  rnn1_kernel<<<PN / 16, 256, 0, stream>>>(event, lengths, Wih1, Whh1, bih1, bhh1,
                                           Wih2, bih2, bhh2, U);
  rnn2_kernel<<<1, HN, 0, stream>>>(U, Whh2, (float*)d_out);
}
